// GATLayer_2671469658544
// MI455X (gfx1250) — matrix operand order repaired and checked by reading, not compiled, not run on hardware
//
#include <hip/hip_runtime.h>
#include <hip/hip_bf16.h>

typedef float  v8f  __attribute__((ext_vector_type(8)));
typedef float  v2f  __attribute__((ext_vector_type(2)));
typedef __bf16 v16bf __attribute__((ext_vector_type(16)));

#define N_NODES 4096
#define IN_F    256
#define OUT_F   512
#define H_HEADS 8
#define D_HEAD  64
#define NEG_SLOPE 0.2f

// ---------------------------------------------------------------------------
// Kernel 1: h = inp @ W.T   (fp32, V_WMMA_F32_16X16X4_F32)
// One wave per 16x16 output tile; K loop over 256 in steps of 4.
// A fragment (16x4 f32): lane L -> row m = L&15, k = (L>>4)*2 + {0,1}
// B fragment (4x16 f32): lane L -> col n = L&15, k = (L>>4)*2 + {0,1}; B[k,n]=W[n,k]
// C/D (16x16 f32): lane L, vgpr r -> row r + (L>>4)*8, col L&15
// ---------------------------------------------------------------------------
__global__ void k_gemm_h(const float* __restrict__ inp,
                         const float* __restrict__ W,
                         float* __restrict__ h) {
    const int wave = threadIdx.x >> 5;
    const int lane = threadIdx.x & 31;
    const int tile = blockIdx.x * 8 + wave;          // 8192 tiles total
    const int ti   = tile / (OUT_F / 16);            // 0..255 row tile
    const int tj   = tile % (OUT_F / 16);            // 0..31  col tile
    const int m    = lane & 15;
    const int half = lane >> 4;

    const float* arow = inp + (size_t)(ti * 16 + m) * IN_F + half * 2;
    const float* brow = W   + (size_t)(tj * 16 + m) * IN_F + half * 2;

    v8f c = {};
    #pragma unroll 4
    for (int k = 0; k < IN_F; k += 4) {
        v2f a = { arow[k], arow[k + 1] };
        v2f b = { brow[k], brow[k + 1] };
        c = __builtin_amdgcn_wmma_f32_16x16x4_f32(false, a, false, b,
                                                  (short)0, c, false, false);
    }

    float* crow = h + (size_t)(ti * 16) * OUT_F + tj * 16 + m;
    #pragma unroll
    for (int r = 0; r < 8; ++r)
        crow[(size_t)(r + half * 8) * OUT_F] = c[r];
}

// ---------------------------------------------------------------------------
// Kernel 2: scores[head][n] = sum_d h[n, head*64+d] * a_left[head*64+d]
// ---------------------------------------------------------------------------
__global__ void k_scores(const float* __restrict__ h,
                         const float* __restrict__ a_left,
                         float* __restrict__ scores) {
    const int idx  = blockIdx.x * blockDim.x + threadIdx.x;  // 0..N*H-1
    const int n    = idx >> 3;
    const int head = idx & 7;
    const float* hp = h + (size_t)n * OUT_F + head * D_HEAD;
    const float* ap = a_left + head * D_HEAD;
    float s = 0.f;
    #pragma unroll 8
    for (int d = 0; d < D_HEAD; ++d) s += hp[d] * ap[d];
    scores[head * N_NODES + n] = s;
}

// ---------------------------------------------------------------------------
// Kernel 3: h_t[head][d][j] = bf16(h[j, head*64+d])   (B-operand layout feed)
// ---------------------------------------------------------------------------
__global__ void k_transpose(const float* __restrict__ h,
                            __bf16* __restrict__ h_t) {
    const int j    = blockIdx.x * 256 + threadIdx.x;  // 0..4095
    const int d    = blockIdx.y;                      // 0..63
    const int head = blockIdx.z;                      // 0..7
    h_t[((size_t)head * D_HEAD + d) * N_NODES + j] =
        (__bf16)h[(size_t)j * OUT_F + head * D_HEAD + d];
}

// ---------------------------------------------------------------------------
// Kernel 4: fused leaky-relu logits + online softmax + P@V (flash-attn style)
// Block = 256 threads (8 waves); each wave owns a 16-row tile of one head.
// blockIdx.x = head*32 + rowblock; rows covered per block = 128.
// P fragment follows ISA 16-bit A-matrix 16x32 layout:
//   lane L: row m = L&15, half = L>>4
//   element e (vgpr v = e>>1, sub = e&1): k = (v<4 ? v*2 : 16+(v-4)*2) + half*8 + sub
// B fragment (32x16 bf16): lane L -> col n = L&15, k = half*8 + e, +8 for e>=8  (8-runs contiguous
//   along j in h_t -> two 16B vector loads per d-tile)
// ---------------------------------------------------------------------------
__global__ void k_attn(const float* __restrict__ A,
                       const float* __restrict__ scores,   // [H][N]
                       const __bf16* __restrict__ h_t,     // [H][64][N]
                       float* __restrict__ out) {          // [N][512]
    __shared__ float s_col[N_NODES];

    const int head = blockIdx.x >> 5;
    const int rb   = blockIdx.x & 31;

    const float* sh = scores + (size_t)head * N_NODES;
    for (int t = threadIdx.x; t < N_NODES; t += 256) s_col[t] = sh[t];
    __syncthreads();

    const int wave = threadIdx.x >> 5;
    const int lane = threadIdx.x & 31;
    const int i0   = rb * 128 + wave * 16;
    const int m    = lane & 15;
    const int half = lane >> 4;

    const float  s_row = s_col[i0 + m];
    const float* Arow  = A + (size_t)(i0 + m) * N_NODES;
    const __bf16* hb   = h_t + (size_t)head * D_HEAD * N_NODES;

    v8f   acc[4] = {{}, {}, {}, {}};
    float row_max = -1e30f;
    float row_sum = 0.f;

    for (int j0 = 0; j0 < N_NODES; j0 += 32) {
        // --- logits for this lane's 16 (row, k) slots ---
        float lg[16];
        float lmax = row_max;
        #pragma unroll
        for (int e = 0; e < 16; ++e) {
            const int v   = e >> 1;
            const int sub = e & 1;
            const int kk  = (v < 4 ? v * 2 : 16 + (v - 4) * 2) + half * 8 + sub;
            const int j   = j0 + kk;
            float x = s_row + s_col[j];
            x = x > 0.f ? x : NEG_SLOPE * x;
            x += Arow[j];
            lg[e] = x;
            lmax = fmaxf(lmax, x);
        }
        // merge the two half-wave copies of each row
        const float m_new = fmaxf(lmax, __shfl_xor(lmax, 16, 32));
        const float alpha = __expf(row_max - m_new);

        // --- P fragment (bf16) + partial row sum ---
        v16bf a;
        float psum = 0.f;
        #pragma unroll
        for (int e = 0; e < 16; ++e) {
            const float p = __expf(lg[e] - m_new);
            psum += p;
            a[e] = (__bf16)p;
        }
        psum += __shfl_xor(psum, 16, 32);
        row_sum = row_sum * alpha + psum;
        row_max = m_new;

        // --- rescale accumulators: acc vgpr r holds row (r + half*8) ---
        #pragma unroll
        for (int r = 0; r < 8; ++r) {
            const float al = __shfl(alpha, r + half * 8, 32);
            #pragma unroll
            for (int dt = 0; dt < 4; ++dt) acc[dt][r] *= al;
        }

        // --- P @ V over the 4 d-tiles ---
        #pragma unroll
        for (int dt = 0; dt < 4; ++dt) {
            const __bf16* bp = hb + (size_t)(dt * 16 + m) * N_NODES + j0 + half * 8;
            typedef __bf16 v8bf __attribute__((ext_vector_type(8)));
            v8bf b_lo = *reinterpret_cast<const v8bf*>(bp);
            v8bf b_hi = *reinterpret_cast<const v8bf*>(bp + 16);
            v16bf b = __builtin_shufflevector(b_lo, b_hi, 0, 1, 2, 3, 4, 5, 6, 7, 8, 9, 10, 11, 12, 13, 14, 15);
            acc[dt] = __builtin_amdgcn_wmma_f32_16x16x32_bf16(
                false, a, false, b, (short)0, acc[dt], false, false);
        }
    }

    // --- normalize and store ---
    #pragma unroll
    for (int r = 0; r < 8; ++r) {
        const float l   = __shfl(row_sum, r + half * 8, 32);
        const float inv = 1.f / l;
        const int   row = i0 + r + half * 8;
        #pragma unroll
        for (int dt = 0; dt < 4; ++dt)
            out[(size_t)row * OUT_F + head * D_HEAD + dt * 16 + m] = acc[dt][r] * inv;
    }
}

// ---------------------------------------------------------------------------
extern "C" void kernel_launch(void* const* d_in, const int* in_sizes, int n_in,
                              void* d_out, int out_size, void* d_ws, size_t ws_size,
                              hipStream_t stream) {
    const float* inp    = (const float*)d_in[0];   // [4096,256]
    const float* A      = (const float*)d_in[1];   // [4096,4096]
    const float* W      = (const float*)d_in[2];   // [512,256]
    const float* a_left = (const float*)d_in[3];   // [1,8,64]
    float* out = (float*)d_out;                    // [4096,512]

    char* ws = (char*)d_ws;
    float*  h_f32  = (float*)ws;                                  // 8 MB
    __bf16* h_t    = (__bf16*)(ws + (size_t)8 * 1024 * 1024);     // 4 MB
    float*  scores = (float*)(ws + (size_t)12 * 1024 * 1024);     // 128 KB

    k_gemm_h   <<<1024, 256, 0, stream>>>(inp, W, h_f32);
    k_scores   <<<(N_NODES * H_HEADS) / 256, 256, 0, stream>>>(h_f32, a_left, scores);
    k_transpose<<<dim3(N_NODES / 256, D_HEAD, H_HEADS), 256, 0, stream>>>(h_f32, h_t);
    k_attn     <<<H_HEADS * 32, 256, 0, stream>>>(A, scores, h_t, out);
}
